// CrossAttention_53283364274522
// MI455X (gfx1250) — compile-verified, matrix operand order checked by reading, not run on hardware
//
#include <hip/hip_runtime.h>

// ---------------------------------------------------------------------------
// CDNA5 (gfx1250) cross-attention pipeline:
//   - all GEMMs + attention matmuls via v_wmma_f32_16x16x32_bf16
//   - workgroup-shared tiles staged into LDS (b128), global_prefetch for next tile
// ---------------------------------------------------------------------------

typedef __attribute__((ext_vector_type(16))) __bf16        v16bf;
typedef __attribute__((ext_vector_type(8)))  float         v8f;
typedef __attribute__((ext_vector_type(8)))  unsigned int  v8u;
typedef __attribute__((ext_vector_type(4)))  unsigned int  u32x4;

static constexpr int BATCH = 2;
static constexpr int NSEQ  = 2048;   // query length
static constexpr int MSEQ  = 2048;   // context length
static constexpr int CDIM  = 1024;
static constexpr int CTXD  = 768;
static constexpr int HEADS = 16;
static constexpr int DH    = 64;

__device__ __forceinline__ unsigned short f2bf(float x) {
  unsigned int u = __builtin_bit_cast(unsigned int, x);
  unsigned int r = u + 0x7FFFu + ((u >> 16) & 1u);   // round-to-nearest-even
  return (unsigned short)(r >> 16);
}

// Load one 16x32 bf16 WMMA fragment (A-layout; B-layout identical on
// transposed storage).  p = start of this lane's row (row = lane&15),
// g = lane>>4.  VGPR v holds K = 16*(v>>2) + 8*g + 2*(v&3) .. +1, so the
// 8 dwords form two 16B-contiguous groups -> compiler merges to b128.
__device__ __forceinline__ v8u frag_ld(const unsigned short* p, int g) {
  v8u f;
#pragma unroll
  for (int v = 0; v < 8; ++v) {
    int k = ((v >> 2) << 4) + (g << 3) + ((v & 3) << 1);
    f[v] = *(const unsigned int*)(p + k);
  }
  return f;
}

__device__ __forceinline__ v8f wmma_bf(v8u a, v8u b, v8f c) {
  return __builtin_amdgcn_wmma_f32_16x16x32_bf16(
      false, __builtin_bit_cast(v16bf, a),
      false, __builtin_bit_cast(v16bf, b),
      (short)0, c, false, false);
}

// ---------------------------------------------------------------------------
// Conversion kernels
// ---------------------------------------------------------------------------
__global__ void k_cvt(const float* __restrict__ src, unsigned short* __restrict__ dst, int n) {
  int i = blockIdx.x * blockDim.x + threadIdx.x;
  if (i < n) dst[i] = f2bf(src[i]);
}

// src [K][N] row-major  ->  dst [N][K] row-major (transpose + convert)
__global__ void k_cvt_t(const float* __restrict__ src, unsigned short* __restrict__ dst,
                        int K, int N) {
  int i = blockIdx.x * blockDim.x + threadIdx.x;
  if (i < K * N) {
    int k = i / N, n = i - k * N;
    dst[(size_t)n * K + k] = f2bf(src[i]);
  }
}

// ---------------------------------------------------------------------------
// bf16 WMMA GEMM:  C[m][n] = sum_k A[m][k] * Bt[n][k]
// Block = 8 waves stacked along M (128 rows x 64 cols). The B tile
// (64 n x 32 k) is shared by all waves -> staged once per k-step into LDS.
// MODE 0: Q (scaled by Dh^-0.5) -> [B,H,N,DH]            (o0)
// MODE 1: KV split -> K [B,H,M,DH] (o0), V^T [B,H,DH,M]  (o1)
// MODE 2: f32 -> outf [row][CDIM]
// ---------------------------------------------------------------------------
template <int MODE>
__global__ __launch_bounds__(256) void k_gemm(
    const unsigned short* __restrict__ A, const unsigned short* __restrict__ Bt,
    float* __restrict__ outf, unsigned short* __restrict__ o0,
    unsigned short* __restrict__ o1, int K) {
  __shared__ unsigned short ldsB[64 * 40];           // 64 rows, stride 40 halves (80B)

  const int tid  = threadIdx.x;
  const int lane = tid & 31;
  const int wave = tid >> 5;
  const int g    = lane >> 4;
  const int c    = lane & 15;
  const int m0   = (blockIdx.y * 8 + wave) * 16;
  const int n0   = blockIdx.x * 64;

  // staging assignment: 64 rows x 4 x b128 = 256 loads, one per thread
  const int br = tid >> 2, bs = (tid & 3) * 8;

  v8f acc[4] = {};
  const unsigned short* arow = A + (size_t)(m0 + c) * K;
  const unsigned short* bstg = Bt + (size_t)(n0 + br) * K + bs;

  for (int k0 = 0; k0 < K; k0 += 32) {
    *(u32x4*)(&ldsB[br * 40 + bs]) = *(const u32x4*)(bstg + k0);
    if (k0 + 32 < K)
      __builtin_prefetch(bstg + k0 + 32, 0, 1);      // global_prefetch_b8
    __syncthreads();

    v8u af = frag_ld(arow + k0, g);
#pragma unroll
    for (int t = 0; t < 4; ++t) {
      v8u bf = frag_ld(&ldsB[(16 * t + c) * 40], g);
      acc[t] = wmma_bf(af, bf, acc[t]);
    }
    __syncthreads();
  }

#pragma unroll
  for (int t = 0; t < 4; ++t) {
#pragma unroll
    for (int r = 0; r < 8; ++r) {
      int row = m0 + r + 8 * g;            // in [0, B*N)
      int col = n0 + 16 * t + c;
      float v = acc[t][r];
      if (MODE == 0) {                      // Q, pre-scaled
        int b = row >> 11, n = row & 2047;
        int h = col >> 6,  d = col & 63;
        o0[(((size_t)(b * HEADS + h) << 11) + n) * DH + d] = f2bf(v * 0.125f);
      } else if (MODE == 1) {               // K / V^T
        int b = row >> 11, m = row & 2047;
        if (col < CDIM) {
          int h = col >> 6, d = col & 63;
          o0[(((size_t)(b * HEADS + h) << 11) + m) * DH + d] = f2bf(v);
        } else {
          int cc = col - CDIM;
          int h = cc >> 6, d = cc & 63;
          o1[(((size_t)(b * HEADS + h) * DH + d) << 11) + m] = f2bf(v);
        }
      } else {                              // f32 output
        outf[(size_t)row * CDIM + col] = v;
      }
    }
  }
}

// ---------------------------------------------------------------------------
// Flash-attention: 8 waves/block, each wave owns a 16-row query tile of the
// same (b,h); the 32-key K/V tiles are staged once per block into LDS.
//   Qh  : [B,H,N,DH]  bf16 (pre-scaled)    Kh : [B,H,M,DH] bf16
//   Vth : [B,H,DH,M]  bf16                 Oout: [B*N, CDIM] bf16
// ---------------------------------------------------------------------------
__global__ __launch_bounds__(256) void k_attn(
    const unsigned short* __restrict__ Qh, const unsigned short* __restrict__ Kh,
    const unsigned short* __restrict__ Vth, unsigned short* __restrict__ Oout) {
  __shared__ unsigned short ldsK[32 * 72];       // 32 keys x 64 d, stride 72 (144B)
  __shared__ unsigned short ldsV[64 * 40];       // 64 d x 32 m,  stride 40 (80B)
  __shared__ unsigned short ldsP[8][16 * 40];    // per-wave 16x32 P tile, stride 40

  const int tid  = threadIdx.x;
  const int lane = tid & 31;
  const int wave = tid >> 5;
  const int g    = lane >> 4;
  const int c    = lane & 15;

  const int bh   = blockIdx.x >> 4;              // 16 tiles of 128 rows per (b,h)
  const int tile = blockIdx.x & 15;
  const int b    = bh >> 4, h = bh & 15;
  const int n0   = tile * 128 + wave * 16;

  const size_t headoff = (size_t)bh * MSEQ * DH;
  const unsigned short* Qb = Qh + headoff + (size_t)(n0 + c) * DH;
  const unsigned short* Kb = Kh + headoff;
  const unsigned short* Vb = Vth + headoff;      // [DH][MSEQ] for this head

  // staging assignment (256 threads):
  const int kr = tid >> 3, ks = (tid & 7) * 8;   // K: 32 rows x 8 segs
  const int vr = tid >> 2, vs = (tid & 3) * 8;   // V: 64 rows x 4 segs
  const unsigned short* kstg = Kb + (size_t)kr * DH + ks;
  const unsigned short* vstg = Vb + (size_t)vr * MSEQ + vs;

  const v8u qa0 = frag_ld(Qb, g);                // d = 0..31
  const v8u qa1 = frag_ld(Qb + 32, g);           // d = 32..63

  float mrow[8], lrow[8];
  v8f o[4] = {};
#pragma unroll
  for (int r = 0; r < 8; ++r) { mrow[r] = -3.4e38f; lrow[r] = 0.f; }

  unsigned short* plds = &ldsP[wave][0];

  for (int m0 = 0; m0 < MSEQ; m0 += 32) {
    // ---- cooperative stage of K/V tiles (b128) + prefetch next tiles ----
    *(u32x4*)(&ldsK[kr * 72 + ks]) = *(const u32x4*)(kstg + (size_t)m0 * DH);
    *(u32x4*)(&ldsV[vr * 40 + vs]) = *(const u32x4*)(vstg + m0);
    if (m0 + 32 < MSEQ) {
      __builtin_prefetch(kstg + (size_t)(m0 + 32) * DH, 0, 1);
      __builtin_prefetch(vstg + m0 + 32, 0, 1);
    }
    __syncthreads();

    // ---- S = Q K^T (16 x 32), f32 accum ----
    v8f s0 = {}, s1 = {};
    {
      const unsigned short* kr0 = &ldsK[c * 72];
      const unsigned short* kr1 = &ldsK[(16 + c) * 72];
      s0 = wmma_bf(qa0, frag_ld(kr0, g), s0);
      s0 = wmma_bf(qa1, frag_ld(kr0 + 32, g), s0);
      s1 = wmma_bf(qa0, frag_ld(kr1, g), s1);
      s1 = wmma_bf(qa1, frag_ld(kr1 + 32, g), s1);
    }

    // ---- online softmax; row r lives in VGPR r across a 16-lane half ----
#pragma unroll
    for (int r = 0; r < 8; ++r) {
      float t = fmaxf(s0[r], s1[r]);
      t = fmaxf(t, __shfl_xor(t, 1, 32));
      t = fmaxf(t, __shfl_xor(t, 2, 32));
      t = fmaxf(t, __shfl_xor(t, 4, 32));
      t = fmaxf(t, __shfl_xor(t, 8, 32));
      float mn = fmaxf(mrow[r], t);
      float alpha = __expf(mrow[r] - mn);
      float p0 = __expf(s0[r] - mn);
      float p1 = __expf(s1[r] - mn);
      float rs = p0 + p1;
      rs += __shfl_xor(rs, 1, 32);
      rs += __shfl_xor(rs, 2, 32);
      rs += __shfl_xor(rs, 4, 32);
      rs += __shfl_xor(rs, 8, 32);
      lrow[r] = lrow[r] * alpha + rs;
      mrow[r] = mn;
      o[0][r] *= alpha; o[1][r] *= alpha; o[2][r] *= alpha; o[3][r] *= alpha;
      int rl = r + 8 * g;                        // C-layout row index
      plds[rl * 40 + c]      = f2bf(p0);         // cols m0   .. m0+15
      plds[rl * 40 + 16 + c] = f2bf(p1);         // cols m0+16.. m0+31
    }

    // make this wave's LDS P-tile visible before fragment reload
    asm volatile("s_wait_dscnt 0" ::: "memory");

    // ---- O += P V  (16 x 64) ----
    v8u pa = frag_ld(plds + c * 40, g);          // A-layout row = lane&15
#pragma unroll
    for (int t = 0; t < 4; ++t) {
      v8u vbf = frag_ld(&ldsV[(16 * t + c) * 40], g);
      o[t] = wmma_bf(pa, vbf, o[t]);
    }
    __syncthreads();                             // before next K/V overwrite
  }

  // ---- normalize + store bf16 to [B*N, CDIM] ----
#pragma unroll
  for (int r = 0; r < 8; ++r) {
    float inv = 1.f / lrow[r];
    size_t row = (size_t)b * NSEQ + n0 + r + 8 * g;
    int colbase = h * DH;
#pragma unroll
    for (int t = 0; t < 4; ++t)
      Oout[row * CDIM + colbase + 16 * t + c] = f2bf(o[t][r] * inv);
  }
}

// ---------------------------------------------------------------------------
// Host-side orchestration
// ---------------------------------------------------------------------------
extern "C" void kernel_launch(void* const* d_in, const int* in_sizes, int n_in,
                              void* d_out, int out_size, void* d_ws, size_t ws_size,
                              hipStream_t stream) {
  const float* x   = (const float*)d_in[0];   // [2,2048,1024]
  const float* ctx = (const float*)d_in[1];   // [2,2048,768]
  const float* Wq  = (const float*)d_in[2];   // [1024,1024]
  const float* Wkv = (const float*)d_in[3];   // [768,2048]
  const float* Wo  = (const float*)d_in[4];   // [1024,1024]
  float* out = (float*)d_out;                 // [2,2048,1024] f32

  unsigned short* ws = (unsigned short*)d_ws;
  const size_t SZ_XB   = (size_t)BATCH * NSEQ * CDIM;
  const size_t SZ_CB   = (size_t)BATCH * MSEQ * CTXD;
  const size_t SZ_WQT  = (size_t)CDIM * CDIM;
  const size_t SZ_WKVT = (size_t)CTXD * 2 * CDIM;
  const size_t SZ_WOT  = (size_t)CDIM * CDIM;
  const size_t SZ_HEAD = (size_t)BATCH * HEADS * MSEQ * DH;

  unsigned short* xb   = ws;
  unsigned short* cb   = xb   + SZ_XB;
  unsigned short* wqt  = cb   + SZ_CB;
  unsigned short* wkvt = wqt  + SZ_WQT;
  unsigned short* wot  = wkvt + SZ_WKVT;
  unsigned short* qws  = wot  + SZ_WOT;
  unsigned short* kws  = qws  + SZ_HEAD;
  unsigned short* vtws = kws  + SZ_HEAD;
  unsigned short* aows = vtws + SZ_HEAD;      // attention out, [B*N, CDIM]

  const int CT = 256;
  k_cvt<<<(int)((SZ_XB + CT - 1) / CT), CT, 0, stream>>>(x, xb, (int)SZ_XB);
  k_cvt<<<(int)((SZ_CB + CT - 1) / CT), CT, 0, stream>>>(ctx, cb, (int)SZ_CB);
  k_cvt_t<<<(int)((SZ_WQT + CT - 1) / CT), CT, 0, stream>>>(Wq, wqt, CDIM, CDIM);
  k_cvt_t<<<(int)((SZ_WKVT + CT - 1) / CT), CT, 0, stream>>>(Wkv, wkvt, CTXD, 2 * CDIM);
  k_cvt_t<<<(int)((SZ_WOT + CT - 1) / CT), CT, 0, stream>>>(Wo, wot, CDIM, CDIM);

  const int MT = BATCH * NSEQ;                // 4096 rows
  k_gemm<0><<<dim3(CDIM / 64, MT / 128), 256, 0, stream>>>(
      xb, wqt, nullptr, qws, nullptr, CDIM);
  k_gemm<1><<<dim3(2 * CDIM / 64, MT / 128), 256, 0, stream>>>(
      cb, wkvt, nullptr, kws, vtws, CTXD);
  k_attn<<<BATCH * HEADS * (NSEQ / 128), 256, 0, stream>>>(qws, kws, vtws, aows);
  k_gemm<2><<<dim3(CDIM / 64, MT / 128), 256, 0, stream>>>(
      aows, wot, out, nullptr, nullptr, CDIM);
}
